// Moondream3DecoderLayer_17978733101590
// MI455X (gfx1250) — matrix operand order repaired and checked by reading, not compiled, not run on hardware
//
#include <hip/hip_runtime.h>
#include <math.h>

// ---------------------------------------------------------------------------
// Moondream3 decoder layer for MI455X (gfx1250, wave32, WMMA + TDM).
//
// Roofline: ~860 MB of fp32 weights stream per call (fc1 536MB + fc2 268MB +
// wqkv/wout 67MB) -> ~37us floor at 23.3 TB/s HBM. FLOPs (~150G) are far
// below the bf16 WMMA ceiling, so the layer is HBM-bound: weights stay fp32
// in memory, the Tensor Data Mover streams 2-D weight/activation tiles into
// double-buffered LDS (s_wait_tensorcnt + barrier sync), and waves convert
// fp32->bf16 via the compiler's native fptrunc feeding
// v_wmma_f32_16x16x32_bf16.  Workspace use: ~285 MB of d_ws.
// ---------------------------------------------------------------------------

#define T_TOK 1024
#define DMODEL 2048
#define NH 32
#define HDIM 64
#define D3 6144
#define NE 32
#define NTOPK 4
#define IDIM 1024
#define ECAP 256
#define LN_EPS 1e-5f

#if __has_builtin(__builtin_amdgcn_tensor_load_to_lds) && \
    __has_builtin(__builtin_amdgcn_s_wait_tensorcnt)
#define MD_HAVE_TDM 1
#else
#define MD_HAVE_TDM 0
#endif

typedef __attribute__((ext_vector_type(16))) __bf16 v16bf;
typedef __attribute__((ext_vector_type(8))) float v8f;
typedef unsigned int uint32x4 __attribute__((ext_vector_type(4)));
typedef int int32x8 __attribute__((ext_vector_type(8)));
typedef int int32x4 __attribute__((ext_vector_type(4)));

union FragU {
  v16bf b;
  uint4 q[2];
  unsigned short s[16];
};

// Native compiler lowering for fp32 -> bf16 (best sequence the target has).
__device__ __forceinline__ unsigned short f2bf(float f) {
  union {
    __bf16 b;
    unsigned short u;
  } c;
  c.b = (__bf16)f;
  return c.u;
}

__device__ __forceinline__ float gelu_exact(float x) {
  return 0.5f * x * (1.f + erff(x * 0.70710678118654752f));
}

__device__ __forceinline__ v8f wmma_bf16(const FragU& a, const FragU& b, v8f c) {
  return __builtin_amdgcn_wmma_f32_16x16x32_bf16(false, a.b, false, b.b,
                                                 (short)0, c, false, false);
}

#if MD_HAVE_TDM
// Issue one TDM 2-D tile load: rows x 32 fp32 elements, row stride = K elems,
// packed contiguously into LDS at lds_off. D# layout per CDNA5 ISA ch.8.
// This toolchain exposes the 6-arg builtin:
//   (uint32x4 g0, int32x8 g1, int32x4 g2, int32x4 g3, int32x8 extra, i32 cpol)
__device__ __forceinline__ void tdm_load_2d(const float* gsrc, unsigned lds_off,
                                            unsigned rows, unsigned stride_elems) {
  unsigned long long ga = (unsigned long long)(const void*)gsrc;
  uint32x4 g0;
  g0[0] = 1u;                                   // count=1, user descriptor
  g0[1] = lds_off;                              // lds_addr (bytes)
  g0[2] = (unsigned)(ga & 0xFFFFFFFFu);         // global_addr lo
  g0[3] = (unsigned)((ga >> 32) & 0x1FFFFFFu)   // global_addr hi
          | (2u << 30);                         // type=2 (image)
  const unsigned td0 = 0x40000000u;             // huge tensor dims: no OOB clip
  const unsigned td1 = 0x40000000u;
  int32x8 g1;
  g1[0] = (int)(2u << 16);                      // wg_mask=0, data_size=4B
  g1[1] = (int)((td0 & 0xFFFFu) << 16);         // atomic_addr=0 | tensor_dim0 lo
  g1[2] = (int)((td0 >> 16) | ((td1 & 0xFFFFu) << 16));
  g1[3] = (int)((td1 >> 16) | (32u << 16));     // tensor_dim1 hi | tile_dim0=32
  g1[4] = (int)(rows);                          // tile_dim1=rows, tile_dim2=0
  g1[5] = (int)stride_elems;                    // tensor_dim0_stride lo32
  g1[6] = 0;                                    // stride0 hi | stride1 lo
  g1[7] = 0;                                    // stride1 hi
  int32x4 gz4 = {0, 0, 0, 0};
  int32x8 gz8 = {0, 0, 0, 0, 0, 0, 0, 0};
  __builtin_amdgcn_tensor_load_to_lds(g0, g1, gz4, gz4, gz8, 0);
}

__device__ __forceinline__ unsigned lds_offset_of(const void* p) {
  return (unsigned)(unsigned long long)p;  // low 32 bits of the pointer as the LDS offset
}
#endif

// Per-wave compute for one 32-wide K step: A tile 64x32, B tile 128x32 (fp32,
// row-major in LDS).  Wave (wm,wn) produces 16 rows x 64 cols in 4 accums.
__device__ __forceinline__ void gemm_compute_step(const float (*Al)[32],
                                                  const float (*Bl)[32], int wm,
                                                  int wn, int lane, v8f* acc) {
  const int r = lane & 15, hi = lane >> 4;
  FragU af;
  {
    const float* ar = Al[wm * 16 + r];
    float4 x0 = *(const float4*)(ar + hi * 16);
    float4 x1 = *(const float4*)(ar + hi * 16 + 4);
    float4 x2 = *(const float4*)(ar + hi * 16 + 8);
    float4 x3 = *(const float4*)(ar + hi * 16 + 12);
    af.b[0] = (__bf16)x0.x;   af.b[1] = (__bf16)x0.y;
    af.b[2] = (__bf16)x0.z;   af.b[3] = (__bf16)x0.w;
    af.b[4] = (__bf16)x1.x;   af.b[5] = (__bf16)x1.y;
    af.b[6] = (__bf16)x1.z;   af.b[7] = (__bf16)x1.w;
    af.b[8] = (__bf16)x2.x;   af.b[9] = (__bf16)x2.y;
    af.b[10] = (__bf16)x2.z;  af.b[11] = (__bf16)x2.w;
    af.b[12] = (__bf16)x3.x;  af.b[13] = (__bf16)x3.y;
    af.b[14] = (__bf16)x3.z;  af.b[15] = (__bf16)x3.w;
  }
#pragma unroll
  for (int bt = 0; bt < 4; ++bt) {
    const float* br = Bl[wn * 64 + bt * 16 + r];
    float4 y0 = *(const float4*)(br + hi * 16);
    float4 y1 = *(const float4*)(br + hi * 16 + 4);
    float4 y2 = *(const float4*)(br + hi * 16 + 8);
    float4 y3 = *(const float4*)(br + hi * 16 + 12);
    FragU bf;
    bf.b[0] = (__bf16)y0.x;   bf.b[1] = (__bf16)y0.y;
    bf.b[2] = (__bf16)y0.z;   bf.b[3] = (__bf16)y0.w;
    bf.b[4] = (__bf16)y1.x;   bf.b[5] = (__bf16)y1.y;
    bf.b[6] = (__bf16)y1.z;   bf.b[7] = (__bf16)y1.w;
    bf.b[8] = (__bf16)y2.x;   bf.b[9] = (__bf16)y2.y;
    bf.b[10] = (__bf16)y2.z;  bf.b[11] = (__bf16)y2.w;
    bf.b[12] = (__bf16)y3.x;  bf.b[13] = (__bf16)y3.y;
    bf.b[14] = (__bf16)y3.z;  bf.b[15] = (__bf16)y3.w;
    acc[bt] = wmma_bf16(af, bf, acc[bt]);
  }
}

// ---------------------------------------------------------------------------
// Generic batched GEMM: C[M,N] = A[M,K] @ B[N,K]^T (+bias).  Block = 256 thr
// (8 waves), tile 64(M)x128(N), KC=32.  TDM double-buffers fp32 tiles in LDS;
// waves convert to bf16 fragments and run 4 WMMA each per K step.
// Requires M%64==0, N%128==0, K%32==0 (true for all call sites).
// ---------------------------------------------------------------------------
__global__ __launch_bounds__(256) void gemm_bf16_kernel(
    const float* __restrict__ A, const float* __restrict__ B,
    const float* __restrict__ bias, float* __restrict__ C, int M, int N, int K,
    long strideA, long strideB, long strideC) {
  A += (long)blockIdx.z * strideA;
  B += (long)blockIdx.z * strideB;
  C += (long)blockIdx.z * strideC;
  const int n0 = blockIdx.x * 128;
  const int m0 = blockIdx.y * 64;
  const int tid = threadIdx.x;
  const int lane = tid & 31;
  const int wave = tid >> 5;
  const int wm = wave & 3;   // M sub-tile (16 rows)
  const int wn = wave >> 2;  // N half (64 cols)

  __shared__ __attribute__((aligned(16))) float Al[2][64][32];
  __shared__ __attribute__((aligned(16))) float Bl[2][128][32];

  v8f acc[4] = {{}, {}, {}, {}};

#if MD_HAVE_TDM
  if (wave == 0) {
    tdm_load_2d(A + (long)m0 * K, lds_offset_of(&Al[0][0][0]), 64, K);
    tdm_load_2d(B + (long)n0 * K, lds_offset_of(&Bl[0][0][0]), 128, K);
  }
  int buf = 0;
  for (int kk = 0; kk < K; kk += 32) {
    if (wave == 0) __builtin_amdgcn_s_wait_tensorcnt(0);
    __syncthreads();  // current buffer is resident for all waves
    if (wave == 0 && kk + 32 < K) {
      tdm_load_2d(A + (long)m0 * K + kk + 32,
                  lds_offset_of(&Al[buf ^ 1][0][0]), 64, K);
      tdm_load_2d(B + (long)n0 * K + kk + 32,
                  lds_offset_of(&Bl[buf ^ 1][0][0]), 128, K);
    }
    gemm_compute_step(Al[buf], Bl[buf], wm, wn, lane, acc);
    buf ^= 1;
  }
#else
  for (int kk = 0; kk < K; kk += 32) {
    {
      const int lr = tid >> 2, lc = (tid & 3) * 8;
      const float4* pa =
          reinterpret_cast<const float4*>(A + (long)(m0 + lr) * K + kk + lc);
      *(float4*)&Al[0][lr][lc] = pa[0];
      *(float4*)&Al[0][lr][lc + 4] = pa[1];
      const int brr = tid >> 1, bc = (tid & 1) * 16;
      const float4* pb =
          reinterpret_cast<const float4*>(B + (long)(n0 + brr) * K + kk + bc);
      __builtin_prefetch(B + (long)(n0 + brr) * K + kk + 32, 0, 1);
      *(float4*)&Bl[0][brr][bc] = pb[0];
      *(float4*)&Bl[0][brr][bc + 4] = pb[1];
      *(float4*)&Bl[0][brr][bc + 8] = pb[2];
      *(float4*)&Bl[0][brr][bc + 12] = pb[3];
    }
    __syncthreads();
    gemm_compute_step(Al[0], Bl[0], wm, wn, lane, acc);
    __syncthreads();
  }
#endif

  // epilogue: C lane layout (lane=col, vgpr=row, hi bit -> row+8)
  const int hi = lane >> 4;
  const int ln = lane & 15;
#pragma unroll
  for (int bt = 0; bt < 4; ++bt) {
#pragma unroll
    for (int vi = 0; vi < 8; ++vi) {
      int row = m0 + wm * 16 + vi + hi * 8;
      int col = n0 + wn * 64 + bt * 16 + ln;
      float bb = bias ? bias[col] : 0.f;
      C[(long)row * N + col] = acc[bt][vi] + bb;
    }
  }
}

// ---------------------------------------------------------------------------
// LayerNorm: h = (x - mu)/sqrt(var+eps)*w + b
// ---------------------------------------------------------------------------
__global__ __launch_bounds__(256) void ln_kernel(const float* __restrict__ x,
                                                 const float* __restrict__ w,
                                                 const float* __restrict__ b,
                                                 float* __restrict__ h) {
  __shared__ float red[256];
  const int t = blockIdx.x, tid = threadIdx.x;
  const float* xr = x + (long)t * DMODEL;
  float s = 0.f;
  for (int i = tid; i < DMODEL; i += 256) s += xr[i];
  red[tid] = s;
  __syncthreads();
  for (int st = 128; st > 0; st >>= 1) {
    if (tid < st) red[tid] += red[tid + st];
    __syncthreads();
  }
  const float mu = red[0] * (1.f / DMODEL);
  __syncthreads();
  float v = 0.f;
  for (int i = tid; i < DMODEL; i += 256) {
    float d = xr[i] - mu;
    v += d * d;
  }
  red[tid] = v;
  __syncthreads();
  for (int st = 128; st > 0; st >>= 1) {
    if (tid < st) red[tid] += red[tid + st];
    __syncthreads();
  }
  const float inv = rsqrtf(red[0] * (1.f / DMODEL) + LN_EPS);
  for (int i = tid; i < DMODEL; i += 256)
    h[(long)t * DMODEL + i] = (xr[i] - mu) * inv * w[i] + b[i];
}

// ---------------------------------------------------------------------------
// tau projections: tok = gelu(qkv); tau_{q,v} = tanh(tok @ w^T) + tau_pos
// ---------------------------------------------------------------------------
__global__ __launch_bounds__(256) void tau_kernel(
    const float* __restrict__ qkv, const float* __restrict__ twq,
    const float* __restrict__ twv, const float* __restrict__ alpha,
    const int* __restrict__ positions, float* __restrict__ tq,
    float* __restrict__ tv) {
  __shared__ float chunk[256];
  const int t = blockIdx.x, tid = threadIdx.x;
  const float* row = qkv + (long)t * D3;
  const int hh = tid >> 1, sel = tid & 1;
  const float* wrow = (sel ? twv : twq) + (long)hh * D3;
  float acc = 0.f;
  for (int base = 0; base < D3; base += 256) {
    chunk[tid] = gelu_exact(row[base + tid]);
    __syncthreads();
    if (tid < 64) {
#pragma unroll 8
      for (int j = 0; j < 256; ++j) acc += chunk[j] * wrow[base + j];
    }
    __syncthreads();
  }
  if (tid < 64) {
    float pl = logf(fmaxf((float)positions[t] + 1.f, 1e-6f));
    float sg = 1.f / (1.f + __expf(-alpha[hh] * pl));
    float val = tanhf(acc) + 1.f + (sg - 0.5f);
    if (sel)
      tv[t * NH + hh] = val;
    else
      tq[t * NH + hh] = val;
  }
}

// ---------------------------------------------------------------------------
// Build bf16 q/k (RoPE'd, tau-scaled q) as [h][t][64], v transposed [h][64][t].
// ---------------------------------------------------------------------------
__global__ __launch_bounds__(64) void qkvh_kernel(
    const float* __restrict__ qkv, const float* __restrict__ tq,
    const float* __restrict__ tv, const int* __restrict__ positions,
    unsigned short* __restrict__ qr, unsigned short* __restrict__ kr,
    unsigned short* __restrict__ vt) {
  const int t = blockIdx.x, h = blockIdx.y, d = threadIdx.x;
  const float* row = qkv + (long)t * D3;
  const float tauq = tq[t * NH + h];
  const float tauv = tv[t * NH + h];
  float vv = row[2 * DMODEL + h * HDIM + d] * tauv;
  float qo = row[h * HDIM + d] * tauq;
  float ko = row[DMODEL + h * HDIM + d];
  if (d < 32) {
    int idx = d & 15;
    float inv_freq = __powf(10000.f, -((float)(2 * idx) / 32.f));
    float ang = (float)positions[t] * inv_freq;
    float c = __cosf(ang), s = __sinf(ang);
    float q1 = row[h * HDIM + idx] * tauq;
    float q2 = row[h * HDIM + idx + 16] * tauq;
    float k1 = row[DMODEL + h * HDIM + idx];
    float k2 = row[DMODEL + h * HDIM + idx + 16];
    if (d < 16) {
      qo = q1 * c - q2 * s;
      ko = k1 * c - k2 * s;
    } else {
      qo = q2 * c + q1 * s;
      ko = k2 * c + k1 * s;
    }
  }
  const long o = ((long)h * T_TOK + t) * HDIM + d;
  qr[o] = f2bf(qo);
  kr[o] = f2bf(ko);
  vt[((long)h * HDIM + d) * T_TOK + t] = f2bf(vv);
}

// ---------------------------------------------------------------------------
// Flash attention: one wave per (head, 16-query tile). QK^T and P.V via WMMA
// bf16; online softmax on f32 accumulators using wave32 shuffles.
// ---------------------------------------------------------------------------
__global__ __launch_bounds__(32) void attn_kernel(
    const unsigned short* __restrict__ qr, const unsigned short* __restrict__ kr,
    const unsigned short* __restrict__ vt, float* __restrict__ ao) {
  const int h = blockIdx.y;
  const int t0 = blockIdx.x * 16;
  const int lane = threadIdx.x;
  const int hi = lane >> 4, ln = lane & 15;

  __shared__ __attribute__((aligned(16))) unsigned short Pl[32][16];

  FragU qf0, qf1;
  {
    const unsigned short* qrow = qr + ((long)h * T_TOK + t0 + ln) * HDIM;
    qf0.q[0] = *reinterpret_cast<const uint4*>(qrow + hi * 16);
    qf0.q[1] = *reinterpret_cast<const uint4*>(qrow + 8 + hi * 16);
    qf1.q[0] = *reinterpret_cast<const uint4*>(qrow + 32 + hi * 16);
    qf1.q[1] = *reinterpret_cast<const uint4*>(qrow + 40 + hi * 16);
  }

  float mrow[8], lrow[8];
  v8f o0 = {}, o1 = {}, o2 = {}, o3 = {};
#pragma unroll
  for (int i = 0; i < 8; ++i) {
    mrow[i] = -__builtin_inff();
    lrow[i] = 0.f;
  }

  const int nsteps = (t0 + 16 + 31) >> 5;
  for (int j = 0; j < nsteps; ++j) {
    const int kv0 = j << 5;
    v8f sA = {}, sB = {};
    {
      const unsigned short* k0 =
          kr + ((long)h * T_TOK + kv0 + ln) * HDIM + hi * 16;
      const unsigned short* k1 =
          kr + ((long)h * T_TOK + kv0 + 16 + ln) * HDIM + hi * 16;
      FragU kf;
      kf.q[0] = *reinterpret_cast<const uint4*>(k0);
      kf.q[1] = *reinterpret_cast<const uint4*>(k0 + 8);
      sA = wmma_bf16(qf0, kf, sA);
      kf.q[0] = *reinterpret_cast<const uint4*>(k0 + 32);
      kf.q[1] = *reinterpret_cast<const uint4*>(k0 + 40);
      sA = wmma_bf16(qf1, kf, sA);
      kf.q[0] = *reinterpret_cast<const uint4*>(k1);
      kf.q[1] = *reinterpret_cast<const uint4*>(k1 + 8);
      sB = wmma_bf16(qf0, kf, sB);
      kf.q[0] = *reinterpret_cast<const uint4*>(k1 + 32);
      kf.q[1] = *reinterpret_cast<const uint4*>(k1 + 40);
      sB = wmma_bf16(qf1, kf, sB);
    }

    float pA[8], pB[8], alpha[8];
#pragma unroll
    for (int vi = 0; vi < 8; ++vi) {
      int rg = t0 + vi + hi * 8;
      float a = sA[vi] * 0.125f;
      float b = sB[vi] * 0.125f;
      if (kv0 + ln > rg) a = -1e30f;
      if (kv0 + 16 + ln > rg) b = -1e30f;
      float lm = fmaxf(a, b);
#pragma unroll
      for (int m = 8; m >= 1; m >>= 1) lm = fmaxf(lm, __shfl_xor(lm, m, 32));
      float mn = fmaxf(mrow[vi], lm);
      float al = __expf(mrow[vi] - mn);
      float ea = __expf(a - mn), eb = __expf(b - mn);
      float ps = ea + eb;
#pragma unroll
      for (int m = 8; m >= 1; m >>= 1) ps += __shfl_xor(ps, m, 32);
      lrow[vi] = lrow[vi] * al + ps;
      mrow[vi] = mn;
      alpha[vi] = al;
      pA[vi] = ea;
      pB[vi] = eb;
    }
#pragma unroll
    for (int vi = 0; vi < 8; ++vi) {
      o0[vi] *= alpha[vi];
      o1[vi] *= alpha[vi];
      o2[vi] *= alpha[vi];
      o3[vi] *= alpha[vi];
    }

#pragma unroll
    for (int vi = 0; vi < 8; ++vi) {
      int row = vi + hi * 8;
      int c0 = ln;
      int c1 = 16 + ln;
      Pl[row + 16 * ((c0 >> 3) & 1)][(c0 & 7) + ((c0 >> 4) << 3)] = f2bf(pA[vi]);
      Pl[row + 16 * ((c1 >> 3) & 1)][(c1 & 7) + ((c1 >> 4) << 3)] = f2bf(pB[vi]);
    }
    __syncthreads();
    FragU pf;
    pf.q[0] = *reinterpret_cast<const uint4*>(&Pl[lane][0]);
    pf.q[1] = *reinterpret_cast<const uint4*>(&Pl[lane][8]);

    const unsigned short* vbase = vt + (long)h * HDIM * T_TOK;
#pragma unroll
    for (int c = 0; c < 4; ++c) {
      const unsigned short* vp =
          vbase + (long)(c * 16 + ln) * T_TOK + kv0 + hi * 8;
      FragU vf;
      vf.q[0] = *reinterpret_cast<const uint4*>(vp);
      vf.q[1] = *reinterpret_cast<const uint4*>(vp + 16);
      if (c == 0) o0 = wmma_bf16(pf, vf, o0);
      if (c == 1) o1 = wmma_bf16(pf, vf, o1);
      if (c == 2) o2 = wmma_bf16(pf, vf, o2);
      if (c == 3) o3 = wmma_bf16(pf, vf, o3);
    }
    __syncthreads();
  }

#pragma unroll
  for (int vi = 0; vi < 8; ++vi) {
    float inv = 1.f / lrow[vi];
    int t = t0 + vi + hi * 8;
    float* orow = ao + (long)t * DMODEL + h * HDIM;
    orow[0 * 16 + ln] = o0[vi] * inv;
    orow[1 * 16 + ln] = o1[vi] * inv;
    orow[2 * 16 + ln] = o2[vi] * inv;
    orow[3 * 16 + ln] = o3[vi] * inv;
  }
}

// ---------------------------------------------------------------------------
// Gating: logits = h @ gate_w^T + gate_b; wave32 top-4; softmax weights.
// ---------------------------------------------------------------------------
__global__ __launch_bounds__(32) void gate_topk_kernel(
    const float* __restrict__ h, const float* __restrict__ gw,
    const float* __restrict__ gb, int* __restrict__ tki,
    float* __restrict__ tkw) {
  const int t = blockIdx.x, e = threadIdx.x;
  const float* hr = h + (long)t * DMODEL;
  const float* wr = gw + (long)e * DMODEL;
  float acc = gb[e];
  for (int i = 0; i < DMODEL; ++i) acc += hr[i] * wr[i];
  float v = acc;
  float vals[4];
  int idxs[4];
#pragma unroll
  for (int k = 0; k < 4; ++k) {
    float bv = v;
    int bi = e;
#pragma unroll
    for (int m = 16; m >= 1; m >>= 1) {
      float ov = __shfl_xor(bv, m, 32);
      int oi = __shfl_xor(bi, m, 32);
      if (ov > bv || (ov == bv && oi < bi)) {
        bv = ov;
        bi = oi;
      }
    }
    vals[k] = bv;
    idxs[k] = bi;
    if (e == bi) v = -__builtin_inff();
  }
  if (e == 0) {
    float mx = vals[0], s = 0.f, w[4];
#pragma unroll
    for (int k = 0; k < 4; ++k) {
      w[k] = __expf(vals[k] - mx);
      s += w[k];
    }
#pragma unroll
    for (int k = 0; k < 4; ++k) {
      tki[t * 4 + k] = idxs[k];
      tkw[t * 4 + k] = w[k] / s;
    }
  }
}

// Expert-parallel ordered slot assignment (matches reference cumsum order).
__global__ __launch_bounds__(32) void slot_kernel(const int* __restrict__ tki,
                                                  int* __restrict__ pos) {
  const int e = threadIdx.x;
  int cnt = 0;
  for (int p = 0; p < T_TOK * NTOPK; ++p)
    if (tki[p] == e) pos[p] = cnt++;
}

// Gather kept tokens into buf[e][slot][:] (buf pre-zeroed).
__global__ __launch_bounds__(256) void scatter_kernel(
    const float* __restrict__ h, const int* __restrict__ tki,
    const int* __restrict__ pos, float* __restrict__ buf) {
  const int p = blockIdx.x;
  const int s = pos[p];
  if (s >= ECAP) return;
  const int e = tki[p];
  const int t = p >> 2;
  const float4* src = reinterpret_cast<const float4*>(h + (long)t * DMODEL);
  float4* dst = reinterpret_cast<float4*>(buf + ((long)e * ECAP + s) * DMODEL);
  for (int i = threadIdx.x; i < DMODEL / 4; i += 256) dst[i] = src[i];
}

// act = gelu(hf[...,:I]) * (hf[...,I:] + 1)
__global__ __launch_bounds__(256) void act_kernel(const float* __restrict__ hf,
                                                  float* __restrict__ act) {
  const long idx = (long)blockIdx.x * 256 + threadIdx.x;
  const long ec = idx / IDIM;
  const long i = idx % IDIM;
  const float* r = hf + ec * (2 * IDIM);
  act[idx] = gelu_exact(r[i]) * (r[IDIM + i] + 1.f);
}

// out = x + attn_out + sum_k w_k * y[e_k][slot_k]
__global__ __launch_bounds__(256) void combine_kernel(
    const float* __restrict__ x, const float* __restrict__ attn,
    const float* __restrict__ y, const int* __restrict__ tki,
    const int* __restrict__ pos, const float* __restrict__ tkw,
    float* __restrict__ out) {
  const int t = blockIdx.x;
  const int d = blockIdx.y * 256 + threadIdx.x;
  const long o = (long)t * DMODEL + d;
  float acc = x[o] + attn[o];
#pragma unroll
  for (int k = 0; k < 4; ++k) {
    int p = t * 4 + k;
    int s = pos[p];
    if (s < ECAP) {
      int e = tki[p];
      acc += tkw[p] * y[((long)e * ECAP + s) * DMODEL + d];
    }
  }
  out[o] = acc;
}

// ---------------------------------------------------------------------------
extern "C" void kernel_launch(void* const* d_in, const int* in_sizes, int n_in,
                              void* d_out, int out_size, void* d_ws,
                              size_t ws_size, hipStream_t stream) {
  (void)in_sizes;
  (void)n_in;
  (void)out_size;
  (void)ws_size;
  const float* x = (const float*)d_in[0];
  const int* positions = (const int*)d_in[1];
  const float* ln_w = (const float*)d_in[2];
  const float* ln_b = (const float*)d_in[3];
  const float* wqkv = (const float*)d_in[4];
  const float* bqkv = (const float*)d_in[5];
  const float* tau_alpha = (const float*)d_in[6];
  const float* tau_wq = (const float*)d_in[7];
  const float* tau_wv = (const float*)d_in[8];
  const float* wout = (const float*)d_in[9];
  const float* bout = (const float*)d_in[10];
  const float* gate_w = (const float*)d_in[11];
  const float* gate_b = (const float*)d_in[12];
  const float* fc1_w = (const float*)d_in[13];
  const float* fc2_w = (const float*)d_in[14];
  float* out = (float*)d_out;

  char* wsp = (char*)d_ws;
  auto balloc = [&](size_t bytes) -> char* {
    char* p = wsp;
    wsp += (bytes + 255) & ~(size_t)255;
    return p;
  };
  float* h = (float*)balloc(sizeof(float) * T_TOK * DMODEL);
  float* qkv = (float*)balloc(sizeof(float) * T_TOK * D3);
  float* tq = (float*)balloc(sizeof(float) * T_TOK * NH);
  float* tv = (float*)balloc(sizeof(float) * T_TOK * NH);
  unsigned short* qr =
      (unsigned short*)balloc(sizeof(unsigned short) * T_TOK * DMODEL);
  unsigned short* kr =
      (unsigned short*)balloc(sizeof(unsigned short) * T_TOK * DMODEL);
  unsigned short* vt =
      (unsigned short*)balloc(sizeof(unsigned short) * T_TOK * DMODEL);
  float* ao = (float*)balloc(sizeof(float) * T_TOK * DMODEL);
  float* attn_out = (float*)balloc(sizeof(float) * T_TOK * DMODEL);
  int* tki = (int*)balloc(sizeof(int) * T_TOK * NTOPK);
  float* tkw = (float*)balloc(sizeof(float) * T_TOK * NTOPK);
  int* pos = (int*)balloc(sizeof(int) * T_TOK * NTOPK);
  float* buf = (float*)balloc(sizeof(float) * NE * ECAP * DMODEL);
  float* hf = (float*)balloc(sizeof(float) * NE * ECAP * 2 * IDIM);
  float* act = (float*)balloc(sizeof(float) * NE * ECAP * IDIM);
  float* y = (float*)balloc(sizeof(float) * NE * ECAP * DMODEL);

  ln_kernel<<<T_TOK, 256, 0, stream>>>(x, ln_w, ln_b, h);

  gemm_bf16_kernel<<<dim3(D3 / 128, T_TOK / 64, 1), 256, 0, stream>>>(
      h, wqkv, bqkv, qkv, T_TOK, D3, DMODEL, 0, 0, 0);

  tau_kernel<<<T_TOK, 256, 0, stream>>>(qkv, tau_wq, tau_wv, tau_alpha,
                                        positions, tq, tv);

  qkvh_kernel<<<dim3(T_TOK, NH), 64, 0, stream>>>(qkv, tq, tv, positions, qr,
                                                  kr, vt);

  attn_kernel<<<dim3(T_TOK / 16, NH), 32, 0, stream>>>(qr, kr, vt, ao);

  gemm_bf16_kernel<<<dim3(DMODEL / 128, T_TOK / 64, 1), 256, 0, stream>>>(
      ao, wout, bout, attn_out, T_TOK, DMODEL, DMODEL, 0, 0, 0);

  gate_topk_kernel<<<T_TOK, 32, 0, stream>>>(h, gate_w, gate_b, tki, tkw);
  slot_kernel<<<1, 32, 0, stream>>>(tki, pos);

  (void)hipMemsetAsync(buf, 0, sizeof(float) * NE * ECAP * DMODEL, stream);
  scatter_kernel<<<T_TOK * NTOPK, 256, 0, stream>>>(h, tki, pos, buf);

  gemm_bf16_kernel<<<dim3(2 * IDIM / 128, ECAP / 64, NE), 256, 0, stream>>>(
      buf, fc1_w, nullptr, hf, ECAP, 2 * IDIM, DMODEL, (long)ECAP * DMODEL,
      (long)2 * IDIM * DMODEL, (long)ECAP * 2 * IDIM);

  act_kernel<<<(NE * ECAP * IDIM) / 256, 256, 0, stream>>>(hf, act);

  gemm_bf16_kernel<<<dim3(DMODEL / 128, ECAP / 64, NE), 256, 0, stream>>>(
      act, fc2_w, nullptr, y, ECAP, DMODEL, IDIM, (long)ECAP * IDIM,
      (long)DMODEL * IDIM, (long)ECAP * DMODEL);

  combine_kernel<<<dim3(T_TOK, DMODEL / 256), 256, 0, stream>>>(
      x, attn_out, y, tki, pos, tkw, out);
}
